// DeepseekV3MoE_52785148067900
// MI455X (gfx1250) — matrix operand order repaired and checked by reading, not compiled, not run on hardware
//
#include <hip/hip_runtime.h>
#include <math.h>

// ---------------------------------------------------------------------------
// DeepseekV3 MoE for MI455X (gfx1250, wave32, WMMA f32_16x16x32_f16).
// Weight-BW-bound: int8 weights dequantized in-register into B fragments.
// Unified 66-expert view: 64 routed + 2 shared halves (I2=1024 = 2 x I=512,
// silu gating is elementwise in i, down-proj additive -> exact split).
// M=32 token tiles: two A fragments reuse each dequantized B fragment,
// halving dequant VALU and weight bytes per FLOP.
// ---------------------------------------------------------------------------

typedef __attribute__((ext_vector_type(16))) _Float16    v16h;
typedef __attribute__((ext_vector_type(8)))  float       v8f;
typedef __attribute__((ext_vector_type(16))) signed char c16;

#define BLKQ   128
#define H_DIM  1024
#define I_DIM  512
#define E_NUM  64
#define EP_NUM 66          // 64 routed + 2 shared halves
#define T_NUM  512
#define K_TOP  6
#define SLOTS  8           // 6 routed + 2 shared slots per token
#define G_NUM  8
#define TG_NUM 3
#define RSCALE 2.5f
#define M_TILE 32

__device__ __forceinline__ float fast_silu(float g) {
  return g * __builtin_amdgcn_rcpf(1.f + __expf(-g));
}

// ---- WMMA fragment helpers (A/B: contiguous k = 16*kh + s; C/D per CDNA5 ISA §7.12.2) ------

// A: 16x32 f16, row-major source [16][lda] in LDS.
__device__ __forceinline__ v16h load_a_frag(const _Float16* __restrict__ A,
                                            int lda, int k0, int lane) {
  const int m  = lane & 15;
  const int kh = (lane >> 4) & 1;
  const _Float16* row = A + m * lda + k0 + 16 * kh;
  v16h a;
#pragma unroll
  for (int v = 0; v < 4; ++v) {        // VGPR v: K = 2v, 2v+1 (+16*kh)
    a[2 * v + 0] = row[2 * v + 0];
    a[2 * v + 1] = row[2 * v + 1];
  }
  row += 8;                             // VGPR 4..7: K = 8 + ...
#pragma unroll
  for (int v = 0; v < 4; ++v) {
    a[8 + 2 * v + 0] = row[2 * v + 0];
    a[8 + 2 * v + 1] = row[2 * v + 1];
  }
  return a;
}

// B: 32x16 f16 from TRANSPOSED weight access: B[k][n] = W[(i0+n)*hd + h0+k].
// Per lane: 16 contiguous int8 (one b128 load), packed-f16 dequant.
__device__ __forceinline__ v16h load_b_frag_wT(const signed char* __restrict__ W,
                                               const float* __restrict__ S,
                                               int i0, int h0, int lane,
                                               int hd, int sld) {
  const int n  = lane & 15;
  const int kh = (lane >> 4) & 1;
  const int row = i0 + n;
  const _Float16 hs = (_Float16)S[(row >> 7) * sld + (h0 >> 7)];
  const c16 wb = *(const c16*)(W + (size_t)row * hd + h0 + 16 * kh);
  v16h b;
#pragma unroll
  for (int i = 0; i < 16; ++i) b[i] = (_Float16)wb[i] * hs;
  return b;
}

// B: 32x16 f16 from NATURAL orientation: B[k][n] = W[(i0+k)*hd + h0+n].
// Cooperative per-wave 32x16 int8 staging in LDS, wave-local s_wait_dscnt.
__device__ __forceinline__ v16h load_b_frag_nat(const signed char* __restrict__ W,
                                                const float* __restrict__ S,
                                                int i0, int h0, int lane,
                                                int hd, int sld,
                                                signed char* __restrict__ stag) {
  const c16 rowb = *(const c16*)(W + (size_t)(i0 + lane) * hd + h0);
  *(c16*)(stag + lane * 16) = rowb;
  __builtin_amdgcn_wave_barrier();
  asm volatile("s_wait_dscnt 0" ::: "memory");
  __builtin_amdgcn_wave_barrier();
  const int n  = lane & 15;
  const int kh = (lane >> 4) & 1;
  const _Float16 hs = (_Float16)S[(i0 >> 7) * sld + ((h0 + n) >> 7)];
  v16h b;
#pragma unroll
  for (int i = 0; i < 16; ++i)
    b[i] = (_Float16)stag[(16 * kh + i) * 16 + n] * hs;
  return b;
}

__device__ __forceinline__ v8f wmma_f16(v16h a, v16h b, v8f c) {
  return __builtin_amdgcn_wmma_f32_16x16x32_f16(false, a, false, b,
                                                (short)0, c, false, false);
}

// ---- kernel 0: init per-expert counters (shared halves take all tokens) ---
__global__ void moe_init_cnt(int* __restrict__ cnt) {
  const int i = threadIdx.x;
  if (i < EP_NUM) cnt[i] = (i < E_NUM) ? 0 : T_NUM;
}

// ---- kernel 1: fill shared-half token lists / weights ---------------------
__global__ void moe_fill_shared(int* __restrict__ list, float* __restrict__ wtbuf) {
  const int idx = blockIdx.x * 256 + threadIdx.x;   // 0 .. 2*T-1
  const int hh = idx >> 9;                          // shared half 0/1
  const int t  = idx & (T_NUM - 1);
  const int tk = t * SLOTS + K_TOP + hh;
  list[(E_NUM + hh) * T_NUM + t] = tk;
  wtbuf[tk] = 1.0f;
}

// ---- kernel 2: router (1 wave per token) ----------------------------------
__global__ __launch_bounds__(32)
void moe_router(const float* __restrict__ x, const float* __restrict__ gate_w,
                int* __restrict__ cnt, int* __restrict__ list,
                float* __restrict__ wtbuf) {
  __shared__ float sc[E_NUM];
  const int t = blockIdx.x;
  const int lane = threadIdx.x;
  const float* xr = x + (size_t)t * H_DIM;
#pragma unroll
  for (int rep = 0; rep < 2; ++rep) {
    const int e = lane + rep * 32;
    const float* gw = gate_w + (size_t)e * H_DIM;
    float acc = 0.f;
    for (int h = 0; h < H_DIM; h += 4) {
      float4 xv = *(const float4*)(xr + h);
      float4 gv = *(const float4*)(gw + h);
      acc += xv.x * gv.x + xv.y * gv.y + xv.z * gv.z + xv.w * gv.w;
    }
    sc[e] = __builtin_amdgcn_rcpf(1.f + __expf(-acc));
  }
  __syncthreads();
  if (lane == 0) {
    const int gsz = E_NUM / G_NUM;
    float gmax[G_NUM];
    bool  gsel[G_NUM];
    for (int g = 0; g < G_NUM; ++g) {
      float m = -1.f;
      for (int j = 0; j < gsz; ++j) m = fmaxf(m, sc[g * gsz + j]);
      gmax[g] = m;
      gsel[g] = false;
    }
    for (int s = 0; s < TG_NUM; ++s) {          // top-TG groups
      int best = 0; float bv = -1e30f;
      for (int g = 0; g < G_NUM; ++g)
        if (!gsel[g] && gmax[g] > bv) { bv = gmax[g]; best = g; }
      gsel[best] = true;
    }
    float masked[E_NUM];
    for (int e = 0; e < E_NUM; ++e)
      masked[e] = gsel[e / gsz] ? sc[e] : -1e30f;
    int   idx[K_TOP]; float wv[K_TOP]; float wsum = 0.f;
    for (int s = 0; s < K_TOP; ++s) {           // top-K experts
      int best = 0; float bv = -2e30f;
      for (int e = 0; e < E_NUM; ++e)
        if (masked[e] > bv) { bv = masked[e]; best = e; }
      idx[s] = best; wv[s] = bv; wsum += bv; masked[best] = -3e30f;
    }
    const float norm = RSCALE / (wsum + 1e-20f);
    for (int s = 0; s < K_TOP; ++s) {
      const int e   = idx[s];
      const int pos = atomicAdd(&cnt[e], 1);
      const int tk  = t * SLOTS + s;
      list[e * T_NUM + pos] = tk;
      wtbuf[tk] = wv[s] * norm;
    }
  }
}

// ---- kernel 3: fused silu-gated MLP over 66 pseudo-experts ----------------
// Block = 256 thr = 8 waves, 32-token tile. Each wave: two A frags (rows
// 0-15 / 16-31) share every dequantized B frag -> 2x B reuse.
__global__ __launch_bounds__(256)
void moe_expert_mlp(const float* __restrict__ x,
                    const signed char* __restrict__ wg, const float* __restrict__ sg,
                    const signed char* __restrict__ wu, const float* __restrict__ su,
                    const signed char* __restrict__ wd, const float* __restrict__ sd,
                    const signed char* __restrict__ sh_wg, const float* __restrict__ sh_sg,
                    const signed char* __restrict__ sh_wu, const float* __restrict__ sh_su,
                    const signed char* __restrict__ sh_wd, const float* __restrict__ sh_sd,
                    const int* __restrict__ cnt, const int* __restrict__ list,
                    const float* __restrict__ wtbuf,
                    float* __restrict__ ybuf) {
  __shared__ _Float16 xA[M_TILE * H_DIM];             // 64 KB
  __shared__ _Float16 hmid[M_TILE * I_DIM];           // 32 KB
  __shared__ signed char stag[8 * 32 * 16];           // 4 KB, per-wave slices
  __shared__ int   orow_s[M_TILE];
  __shared__ float wt_s[M_TILE];

  const int tid  = threadIdx.x;
  const int lane = tid & 31;
  const int wave = tid >> 5;

  const int e    = blockIdx.x;                        // 0..65
  const int tile = blockIdx.y;
  const int c    = cnt[e];
  if (tile * M_TILE >= c) return;
  const int nrows = min(M_TILE, c - tile * M_TILE);

  const size_t wsz = (size_t)I_DIM * H_DIM;
  const size_t ssz = (size_t)(I_DIM / BLKQ) * (H_DIM / BLKQ);
  const signed char *Wg, *Wu, *Wd;
  const float *Sg, *Su, *Sd;
  if (e < E_NUM) {
    Wg = wg + (size_t)e * wsz;  Sg = sg + (size_t)e * ssz;
    Wu = wu + (size_t)e * wsz;  Su = su + (size_t)e * ssz;
    Wd = wd + (size_t)e * wsz;  Sd = sd + (size_t)e * ssz;
  } else {
    const int hh = e - E_NUM;                         // shared half 0/1
    Wg = sh_wg + (size_t)hh * wsz;  Sg = sh_sg + (size_t)hh * ssz;
    Wu = sh_wu + (size_t)hh * wsz;  Su = sh_su + (size_t)hh * ssz;
    Wd = sh_wd + (size_t)hh * wsz;  Sd = sh_sd + (size_t)hh * ssz;
  }

  __shared__ int tok_s[M_TILE];
  if (tid < M_TILE) {
    const int r  = min(tid, nrows - 1);               // clamp: dup last row
    const int tk = list[e * T_NUM + tile * M_TILE + r];
    tok_s[tid]  = tk / SLOTS;
    orow_s[tid] = tk;
    wt_s[tid]   = wtbuf[tk];
  }
  __syncthreads();

  // ---- gather x tile -> LDS f16 [32][H] (each thread: 128 elems) ----
  {
    const int r  = tid >> 3;                          // 0..31
    const int c0 = (tid & 7) * 128;
    const float* src = x + (size_t)tok_s[r] * H_DIM + c0;
    _Float16* dst = xA + r * H_DIM + c0;
#pragma unroll
    for (int i = 0; i < 128; i += 4) {
      const float4 v = *(const float4*)(src + i);
      dst[i + 0] = (_Float16)v.x; dst[i + 1] = (_Float16)v.y;
      dst[i + 2] = (_Float16)v.z; dst[i + 3] = (_Float16)v.w;
    }
  }
  __syncthreads();

  // ---- phase 1: g = x*Wg^T, u = x*Wu^T, hmid = silu(g)*u ----
  for (int nt = 0; nt < (I_DIM / 16) / 8; ++nt) {
    const int i0 = (wave + nt * 8) * 16;
    v8f cg0 = {}, cg1 = {}, cu0 = {}, cu1 = {};
    for (int k0 = 0; k0 < H_DIM; k0 += 32) {
      const v16h a0 = load_a_frag(xA, H_DIM, k0, lane);
      const v16h a1 = load_a_frag(xA + 16 * H_DIM, H_DIM, k0, lane);
      const v16h bg = load_b_frag_wT(Wg, Sg, i0, k0, lane, H_DIM, H_DIM / BLKQ);
      const v16h bu = load_b_frag_wT(Wu, Su, i0, k0, lane, H_DIM, H_DIM / BLKQ);
      cg0 = wmma_f16(a0, bg, cg0);
      cg1 = wmma_f16(a1, bg, cg1);
      cu0 = wmma_f16(a0, bu, cu0);
      cu1 = wmma_f16(a1, bu, cu1);
    }
    const int n  = lane & 15;          // C/D: lane 0-15 -> M=v, 16-31 -> M=8+v
    const int mb = (lane >> 4) * 8;
#pragma unroll
    for (int v = 0; v < 8; ++v) {
      hmid[(mb + v) * I_DIM + i0 + n]      = (_Float16)(fast_silu(cg0[v]) * cu0[v]);
      hmid[(16 + mb + v) * I_DIM + i0 + n] = (_Float16)(fast_silu(cg1[v]) * cu1[v]);
    }
  }
  __syncthreads();

  // ---- phase 2: y = hmid * Wd, per-row weighted scatter ----
  signed char* mystag = stag + wave * 32 * 16;
  for (int nt = 0; nt < (H_DIM / 16) / 8; ++nt) {
    const int h0 = (wave + nt * 8) * 16;
    v8f c0 = {}, c1 = {};
    for (int i0 = 0; i0 < I_DIM; i0 += 32) {
      if (i0 + 32 < I_DIM)                            // prefetch next chunk
        __builtin_prefetch(Wd + (size_t)(i0 + 32 + lane) * H_DIM + h0, 0, 1);
      const v16h a0 = load_a_frag(hmid, I_DIM, i0, lane);
      const v16h a1 = load_a_frag(hmid + 16 * I_DIM, I_DIM, i0, lane);
      const v16h b  = load_b_frag_nat(Wd, Sd, i0, h0, lane, H_DIM,
                                      H_DIM / BLKQ, mystag);
      c0 = wmma_f16(a0, b, c0);
      c1 = wmma_f16(a1, b, c1);
    }
    const int n  = lane & 15;
    const int mb = (lane >> 4) * 8;
#pragma unroll
    for (int v = 0; v < 8; ++v) {
      const int m0 = mb + v, m1 = 16 + mb + v;
      if (m0 < nrows)
        ybuf[(size_t)orow_s[m0] * H_DIM + h0 + n] = c0[v] * wt_s[m0];
      if (m1 < nrows)
        ybuf[(size_t)orow_s[m1] * H_DIM + h0 + n] = c1[v] * wt_s[m1];
    }
  }
}

// ---- kernel 4: combine all 8 slots per token ------------------------------
__global__ __launch_bounds__(256)
void moe_combine(const float* __restrict__ ybuf, float* __restrict__ out) {
  const int i = blockIdx.x * 256 + threadIdx.x;       // over T*H
  const int t = i / H_DIM, h = i - t * H_DIM;
  float acc = 0.f;
#pragma unroll
  for (int k = 0; k < SLOTS; ++k)
    acc += ybuf[(size_t)(t * SLOTS + k) * H_DIM + h];
  out[i] = acc;
}

// ---------------------------------------------------------------------------
extern "C" void kernel_launch(void* const* d_in, const int* in_sizes, int n_in,
                              void* d_out, int out_size, void* d_ws,
                              size_t ws_size, hipStream_t stream) {
  (void)in_sizes; (void)n_in; (void)out_size; (void)ws_size;
  const float*       x      = (const float*)d_in[0];
  const float*       gate_w = (const float*)d_in[1];
  const signed char* wg = (const signed char*)d_in[2];
  const float*       sg = (const float*)d_in[3];
  const signed char* wu = (const signed char*)d_in[4];
  const float*       su = (const float*)d_in[5];
  const signed char* wd = (const signed char*)d_in[6];
  const float*       sd = (const float*)d_in[7];
  const signed char* sh_wg = (const signed char*)d_in[8];
  const float*       sh_sg = (const float*)d_in[9];
  const signed char* sh_wu = (const signed char*)d_in[10];
  const float*       sh_su = (const float*)d_in[11];
  const signed char* sh_wd = (const signed char*)d_in[12];
  const float*       sh_sd = (const float*)d_in[13];
  float* out = (float*)d_out;

  // workspace layout (~16.2 MB)
  uint8_t* ws = (uint8_t*)d_ws;
  int*   cnt   = (int*)ws;              ws += 256;
  int*   list  = (int*)ws;              ws += (size_t)EP_NUM * T_NUM * 4;
  float* wtbuf = (float*)ws;            ws += (size_t)T_NUM * SLOTS * 4;
  float* ybuf  = (float*)ws;            // [T*SLOTS, H] f32

  moe_init_cnt<<<1, 128, 0, stream>>>(cnt);
  moe_fill_shared<<<(2 * T_NUM) / 256, 256, 0, stream>>>(list, wtbuf);
  moe_router<<<T_NUM, 32, 0, stream>>>(x, gate_w, cnt, list, wtbuf);
  moe_expert_mlp<<<dim3(EP_NUM, T_NUM / M_TILE), 256, 0, stream>>>(
      x, wg, sg, wu, su, wd, sd,
      sh_wg, sh_sg, sh_wu, sh_su, sh_wd, sh_sd,
      cnt, list, wtbuf, ybuf);
  moe_combine<<<(T_NUM * H_DIM) / 256, 256, 0, stream>>>(ybuf, out);
}
